// GCNModel_16518444221032
// MI455X (gfx1250) — matrix operand order repaired and checked by reading, not compiled, not run on hardware
//
#include <hip/hip_runtime.h>
#include <hip/hip_bf16.h>
#include <math.h>

typedef __attribute__((ext_vector_type(16))) _Float16 v16h;
typedef __attribute__((ext_vector_type(8)))  float    v8f;

#define N_    50000
#define E_    1600000
#define P_    500000
#define INDIM 512
#define HID   128
#define OUTF  64

union FragB { v16h h; uint4 q[2]; unsigned u[8]; };

// ---------------------------------------------------------------------------
__global__ void zero_kernel(unsigned* __restrict__ p, int n) {
    int t = blockIdx.x * blockDim.x + threadIdx.x;
    if (t < n) p[t] = 0u;
}

__global__ void deg_kernel(const int* __restrict__ ei, int* __restrict__ deg) {
    int e = blockIdx.x * blockDim.x + threadIdx.x;
    if (e < E_) atomicAdd(&deg[ei[E_ + e]], 1);
}

__global__ void dinv_kernel(const int* __restrict__ deg, float* __restrict__ dinv) {
    int i = blockIdx.x * blockDim.x + threadIdx.x;
    if (i < N_) dinv[i] = rsqrtf((float)deg[i] + 1.0f);
}

// ---------------------------------------------------------------------------
// Pack fp32 weights [K][NC] row-major into lane-contiguous WMMA B layout:
//   pk[((ks*tilesN + tn)*32 + lane)*8 + v] = {f16 W[k0][n], f16 W[k0+1][n]}
//   k0 = ks*32 + (v>>2)*16 + (lane>>4)*8 + 2*(v&3) ; n = tn*16 + (lane&15)
__global__ void pack_w_kernel(const float* __restrict__ W, unsigned* __restrict__ pk,
                              int K, int NC) {
    int t = blockIdx.x * blockDim.x + threadIdx.x;
    int tilesN = NC >> 4;
    int total = (K >> 5) * tilesN * 256;
    if (t >= total) return;
    int lane = t & 31;
    int v    = (t >> 5) & 7;
    int tn   = (t >> 8) % tilesN;
    int ks   = t / (256 * tilesN);
    int k0   = ks * 32 + (v >> 2) * 16 + ((lane >> 4) & 1) * 8 + 2 * (v & 3);
    int n    = tn * 16 + (lane & 15);
    union { _Float16 h[2]; unsigned u; } c;
    c.h[0] = (_Float16)W[(size_t)k0 * NC + n];
    c.h[1] = (_Float16)W[(size_t)(k0 + 1) * NC + n];
    pk[(((size_t)(ks * tilesN + tn) * 32) + lane) * 8 + v] = c.u;
}

// ---------------------------------------------------------------------------
// GEMM1: xw = x[50000x512] @ W1[512x128]; one wave -> 16 rows x 128 cols.
// Entire packed W1 (128KB) staged in LDS once per block; B tiles then feed
// the WMMAs via ds_load_b128 instead of per-wave L2 round-trips.
#define W1P_DW ((INDIM / 32) * (HID / 16) * 256)   // 32768 dwords = 128KB
__global__ void __launch_bounds__(256) gemm1_kernel(const float* __restrict__ x,
                                                    const unsigned* __restrict__ w1p,
                                                    float* __restrict__ xw) {
    extern __shared__ unsigned lds[];
    #pragma unroll
    for (int j = 0; j < W1P_DW / (256 * 4); ++j) {
        int i = (j * 256 + threadIdx.x) * 4;
        *(uint4*)&lds[i] = *(const uint4*)&w1p[i];
    }
    __syncthreads();

    int wave = (blockIdx.x * blockDim.x + threadIdx.x) >> 5;
    if (wave >= N_ / 16) return;
    int lane = threadIdx.x & 31;
    int m = lane & 15, half = lane >> 4;
    const float* xrow = x + (size_t)(wave * 16 + m) * INDIM;
    v8f acc[8] = {};
    for (int ks = 0; ks < INDIM / 32; ++ks) {
        // A fragment: two contiguous 32B chunks of the row, f32 -> f16
        const float* ap = xrow + ks * 32 + half * 8;
        float4 f0 = *(const float4*)(ap);
        float4 f1 = *(const float4*)(ap + 4);
        float4 f2 = *(const float4*)(ap + 16);
        float4 f3 = *(const float4*)(ap + 20);
        v16h a;
        a[0]  = (_Float16)f0.x; a[1]  = (_Float16)f0.y;
        a[2]  = (_Float16)f0.z; a[3]  = (_Float16)f0.w;
        a[4]  = (_Float16)f1.x; a[5]  = (_Float16)f1.y;
        a[6]  = (_Float16)f1.z; a[7]  = (_Float16)f1.w;
        a[8]  = (_Float16)f2.x; a[9]  = (_Float16)f2.y;
        a[10] = (_Float16)f2.z; a[11] = (_Float16)f2.w;
        a[12] = (_Float16)f3.x; a[13] = (_Float16)f3.y;
        a[14] = (_Float16)f3.z; a[15] = (_Float16)f3.w;
        #pragma unroll
        for (int tn = 0; tn < 8; ++tn) {
            const unsigned* bp = lds + (((ks * 8 + tn) * 32) + lane) * 8;
            FragB b;
            b.q[0] = *(const uint4*)(bp);
            b.q[1] = *(const uint4*)(bp + 4);
            acc[tn] = __builtin_amdgcn_wmma_f32_16x16x32_f16(
                false, a, false, b.h, (short)0, acc[tn], false, false);
        }
    }
    float* orow = xw + (size_t)wave * 16 * HID;
    #pragma unroll
    for (int tn = 0; tn < 8; ++tn)
        #pragma unroll
        for (int r = 0; r < 8; ++r)
            orow[(size_t)(r + half * 8) * HID + tn * 16 + m] = acc[tn][r];
}

// GEMM2: xw2 = h16[50000x128] @ W2[128x64]; packed W2 (16KB) staged in LDS
#define W2P_DW ((HID / 32) * (OUTF / 16) * 256)    // 4096 dwords = 16KB
__global__ void __launch_bounds__(256) gemm2_kernel(const _Float16* __restrict__ h16,
                                                    const unsigned* __restrict__ w2p,
                                                    float* __restrict__ out) {
    extern __shared__ unsigned lds[];
    #pragma unroll
    for (int j = 0; j < W2P_DW / (256 * 4); ++j) {
        int i = (j * 256 + threadIdx.x) * 4;
        *(uint4*)&lds[i] = *(const uint4*)&w2p[i];
    }
    __syncthreads();

    int wave = (blockIdx.x * blockDim.x + threadIdx.x) >> 5;
    if (wave >= N_ / 16) return;
    int lane = threadIdx.x & 31;
    int m = lane & 15, half = lane >> 4;
    const unsigned* arow = (const unsigned*)(h16 + (size_t)(wave * 16 + m) * HID);
    v8f acc[4] = {};
    #pragma unroll
    for (int ks = 0; ks < HID / 32; ++ks) {
        int base = ks * 16 + half * 4;       // pair-index of first A dword
        FragB a;
        a.q[0] = *(const uint4*)(arow + base);
        a.q[1] = *(const uint4*)(arow + base + 8);
        #pragma unroll
        for (int tn = 0; tn < 4; ++tn) {
            const unsigned* bp = lds + (((ks * 4 + tn) * 32) + lane) * 8;
            FragB b;
            b.q[0] = *(const uint4*)(bp);
            b.q[1] = *(const uint4*)(bp + 4);
            acc[tn] = __builtin_amdgcn_wmma_f32_16x16x32_f16(
                false, a.h, false, b.h, (short)0, acc[tn], false, false);
        }
    }
    float* orow = out + (size_t)wave * 16 * OUTF;
    #pragma unroll
    for (int tn = 0; tn < 4; ++tn)
        #pragma unroll
        for (int r = 0; r < 8; ++r)
            orow[(size_t)(r + half * 8) * OUTF + tn * 16 + m] = acc[tn][r];
}

// ---------------------------------------------------------------------------
// edge scatter-add: agg[col] += dinv[row]*dinv[col] * feat[row], float4 granules
__global__ void agg_kernel(const int* __restrict__ ei, const float* __restrict__ dinv,
                           const float* __restrict__ feat, float* __restrict__ agg,
                           int F) {
    int t = blockIdx.x * blockDim.x + threadIdx.x;
    int chunks = F >> 2;
    if (t >= E_ * chunks) return;
    int e = t / chunks;
    int f = (t % chunks) * 4;
    int r = ei[e], c = ei[E_ + e];
    float norm = dinv[r] * dinv[c];
    const float4 v = *(const float4*)(feat + (size_t)r * F + f);
    float* a = agg + (size_t)c * F + f;
    atomicAdd(a + 0, norm * v.x);
    atomicAdd(a + 1, norm * v.y);
    atomicAdd(a + 2, norm * v.z);
    atomicAdd(a + 3, norm * v.w);
}

// fused self-loop + bias + BN(eval) + ReLU -> f16 activations
__global__ void post_kernel(const float* __restrict__ agg, const float* __restrict__ xw,
                            const float* __restrict__ dinv, const float* __restrict__ b,
                            const float* __restrict__ gamma, const float* __restrict__ beta,
                            const float* __restrict__ mean, const float* __restrict__ var,
                            _Float16* __restrict__ o16, int F, int logF) {
    int t = blockIdx.x * blockDim.x + threadIdx.x;
    if (t >= N_ * F) return;
    int i = t >> logF, f = t & (F - 1);
    float di = dinv[i];
    float val = agg[t] + di * di * xw[t] + b[f];
    val = (val - mean[f]) * rsqrtf(var[f] + 1e-5f) * gamma[f] + beta[f];
    o16[t] = (_Float16)fmaxf(val, 0.0f);
}

// ---------------------------------------------------------------------------
// edge head: A = [z16[src[p]] ‖ z16[dst[p]]] (16 pairs/wave, K=128) @ hW1,
// fused +hb1, ReLU, dot(hW2), +hb2, sigmoid via in-wave shuffle reduce.
// Packed hW1 (16KB) staged in LDS; each 32-wide k-step is all-src or all-dst.
__global__ void __launch_bounds__(256) head_kernel(const _Float16* __restrict__ z16,
                            const int* __restrict__ src, const int* __restrict__ dst,
                            const unsigned* __restrict__ hw1p,
                            const float* __restrict__ hb1, const float* __restrict__ hW2,
                            const float* __restrict__ hb2, float* __restrict__ out) {
    extern __shared__ unsigned lds[];
    #pragma unroll
    for (int j = 0; j < W2P_DW / (256 * 4); ++j) {
        int i = (j * 256 + threadIdx.x) * 4;
        *(uint4*)&lds[i] = *(const uint4*)&hw1p[i];
    }
    __syncthreads();

    int wave = (blockIdx.x * blockDim.x + threadIdx.x) >> 5;
    if (wave >= P_ / 16) return;
    int lane = threadIdx.x & 31;
    int m = lane & 15, half = lane >> 4;
    int p = wave * 16 + m;
    const unsigned* zs = (const unsigned*)(z16 + (size_t)src[p] * OUTF);
    const unsigned* zd = (const unsigned*)(z16 + (size_t)dst[p] * OUTF);
    v8f acc[4] = {};
    #pragma unroll
    for (int ks = 0; ks < 4; ++ks) {
        const unsigned* zr = (ks < 2) ? zs : zd;
        int base = (ks & 1) * 16 + half * 4;  // pair-index within the 64-col half
        FragB a;
        a.q[0] = *(const uint4*)(zr + base);
        a.q[1] = *(const uint4*)(zr + base + 8);
        #pragma unroll
        for (int tn = 0; tn < 4; ++tn) {
            const unsigned* bp = lds + (((ks * 4 + tn) * 32) + lane) * 8;
            FragB b;
            b.q[0] = *(const uint4*)(bp);
            b.q[1] = *(const uint4*)(bp + 4);
            acc[tn] = __builtin_amdgcn_wmma_f32_16x16x32_f16(
                false, a.h, false, b.h, (short)0, acc[tn], false, false);
        }
    }
    float s[8] = {0.f, 0.f, 0.f, 0.f, 0.f, 0.f, 0.f, 0.f};
    #pragma unroll
    for (int tn = 0; tn < 4; ++tn) {
        int n = tn * 16 + m;
        float bias = hb1[n], w = hW2[n];
        #pragma unroll
        for (int r = 0; r < 8; ++r) {
            float hm = fmaxf(acc[tn][r] + bias, 0.0f);
            s[r] += hm * w;
        }
    }
    float hb = hb2[0];
    #pragma unroll
    for (int r = 0; r < 8; ++r) {
        float v = s[r];
        v += __shfl_xor(v, 1, 32);
        v += __shfl_xor(v, 2, 32);
        v += __shfl_xor(v, 4, 32);
        v += __shfl_xor(v, 8, 32);
        if (m == 0) {
            float logit = v + hb;
            out[wave * 16 + r + half * 8] = 1.0f / (1.0f + expf(-logit));
        }
    }
}

// ---------------------------------------------------------------------------
extern "C" void kernel_launch(void* const* d_in, const int* in_sizes, int n_in,
                              void* d_out, int out_size, void* d_ws, size_t ws_size,
                              hipStream_t stream) {
    const float* x     = (const float*)d_in[0];
    const int*   ei    = (const int*)d_in[1];
    const int*   src   = (const int*)d_in[2];
    const int*   dst   = (const int*)d_in[3];
    const float* W1    = (const float*)d_in[4];
    const float* b1    = (const float*)d_in[5];
    const float* bn1_g = (const float*)d_in[6];
    const float* bn1_b = (const float*)d_in[7];
    const float* bn1_m = (const float*)d_in[8];
    const float* bn1_v = (const float*)d_in[9];
    const float* W2    = (const float*)d_in[10];
    const float* b2    = (const float*)d_in[11];
    const float* bn2_g = (const float*)d_in[12];
    const float* bn2_b = (const float*)d_in[13];
    const float* bn2_m = (const float*)d_in[14];
    const float* bn2_v = (const float*)d_in[15];
    const float* hW1   = (const float*)d_in[16];
    const float* hb1   = (const float*)d_in[17];
    const float* hW2   = (const float*)d_in[18];
    const float* hb2   = (const float*)d_in[19];
    float* out = (float*)d_out;

    // workspace layout (dwords)
    float* ws = (float*)d_ws;
    size_t o = 0;
    int*       deg  = (int*)(ws + o);       o += N_;
    float*     dinv = ws + o;               o += N_;
    float*     xw1  = ws + o;               o += (size_t)N_ * HID;
    float*     agg1 = ws + o;               o += (size_t)N_ * HID;
    _Float16*  h16  = (_Float16*)(ws + o);  o += (size_t)N_ * HID / 2;
    float*     xw2  = ws + o;               o += (size_t)N_ * OUTF;
    float*     agg2 = ws + o;               o += (size_t)N_ * OUTF;
    _Float16*  z16  = (_Float16*)(ws + o);  o += (size_t)N_ * OUTF / 2;
    unsigned*  w1p  = (unsigned*)(ws + o);  o += W1P_DW;
    unsigned*  w2p  = (unsigned*)(ws + o);  o += W2P_DW;
    unsigned*  hw1p = (unsigned*)(ws + o);  o += W2P_DW;

    // allow 128KB dynamic LDS for gemm1 (host-side attribute, capture-safe)
    hipFuncSetAttribute((const void*)gemm1_kernel,
                        hipFuncAttributeMaxDynamicSharedMemorySize, W1P_DW * 4);

    const int B = 256;
    // zero deg + both agg buffers (span also covers dinv/xw1/h16/xw2, all of
    // which are fully overwritten later in the pipeline)
    int nz = N_ * (2 + HID + HID + HID / 2 + OUTF + OUTF);
    zero_kernel<<<(nz + B - 1) / B, B, 0, stream>>>((unsigned*)d_ws, nz);

    deg_kernel<<<(E_ + B - 1) / B, B, 0, stream>>>(ei, deg);
    dinv_kernel<<<(N_ + B - 1) / B, B, 0, stream>>>(deg, dinv);

    pack_w_kernel<<<W1P_DW / B, B, 0, stream>>>(W1, w1p, INDIM, HID);
    pack_w_kernel<<<W2P_DW / B, B, 0, stream>>>(W2, w2p, HID, OUTF);
    pack_w_kernel<<<W2P_DW / B, B, 0, stream>>>(hW1, hw1p, HID, OUTF);

    // layer 1
    gemm1_kernel<<<((N_ / 16) * 32 + B - 1) / B, B, W1P_DW * 4, stream>>>(x, w1p, xw1);
    agg_kernel<<<(E_ * (HID / 4) + B - 1) / B, B, 0, stream>>>(ei, dinv, xw1, agg1, HID);
    post_kernel<<<(N_ * HID + B - 1) / B, B, 0, stream>>>(agg1, xw1, dinv, b1, bn1_g, bn1_b,
                                                          bn1_m, bn1_v, h16, HID, 7);
    // layer 2
    gemm2_kernel<<<((N_ / 16) * 32 + B - 1) / B, B, W2P_DW * 4, stream>>>(h16, w2p, xw2);
    agg_kernel<<<(E_ * (OUTF / 4) + B - 1) / B, B, 0, stream>>>(ei, dinv, xw2, agg2, OUTF);
    post_kernel<<<(N_ * OUTF + B - 1) / B, B, 0, stream>>>(agg2, xw2, dinv, b2, bn2_g, bn2_b,
                                                           bn2_m, bn2_v, z16, OUTF, 6);
    // edge head
    head_kernel<<<((P_ / 16) * 32 + B - 1) / B, B, W2P_DW * 4, stream>>>(z16, src, dst, hw1p,
                                                                         hb1, hW2, hb2, out);
}
